// Decoder_773094113440
// MI455X (gfx1250) — matrix operand order repaired and checked by reading, not compiled, not run on hardware
//
#include <hip/hip_runtime.h>
#include <math.h>

// ---------------- problem constants ----------------
#define B_SZ   128
#define HID    2048
#define GATES  8192            // 4*HID
#define LAT    128
#define KX     160             // 133 padded up to multiple of 32
#define NOUT   63
#define NOUTP  64              // padded
#define NMAX   10
#define SQRT_T 0.70710678118654752f

typedef _Float16 v16h __attribute__((ext_vector_type(16)));
typedef _Float16 v8h  __attribute__((ext_vector_type(8)));
typedef _Float16 v4h  __attribute__((ext_vector_type(4)));
typedef float    v8f  __attribute__((ext_vector_type(8)));
typedef float    v4f  __attribute__((ext_vector_type(4)));

#define WMMA_F16(a, b, c) \
  __builtin_amdgcn_wmma_f32_16x16x32_f16(false, (a), false, (b), (short)0, (c), false, false)

// ---------------- async global->LDS path (guarded; falls back to reg copy) ---
#if defined(__HIP_DEVICE_COMPILE__) && __has_builtin(__builtin_amdgcn_global_load_async_to_lds_b128)
#define HAVE_ASYNC_LDS 1
#else
#define HAVE_ASYNC_LDS 0
#endif

#if HAVE_ASYNC_LDS
// Builtin expects pointers to int4 (GCC vector_size style) in AS1 (global,
// printed by clang as "__device__") and AS3 (LDS, "__shared__").
typedef int v4i_vs __attribute__((vector_size(16)));
typedef __attribute__((address_space(1))) v4i_vs as1_v4i;
typedef __attribute__((address_space(3))) v4i_vs as3_v4i;
#endif

// copy 16 bytes (8 halves) global -> LDS
__device__ __forceinline__ void cp16_g2l(const _Float16* g, _Float16* l) {
#if HAVE_ASYNC_LDS
  __builtin_amdgcn_global_load_async_to_lds_b128((as1_v4i*)g, (as3_v4i*)l, 0, 0);
#else
  *(v8h*)l = *(const v8h*)g;
#endif
}

__device__ __forceinline__ void async_wait0() {
#if HAVE_ASYNC_LDS
#if __has_builtin(__builtin_amdgcn_s_wait_asynccnt)
  __builtin_amdgcn_s_wait_asynccnt(0);
#else
  asm volatile("s_wait_asynccnt 0" ::: "memory");
#endif
#endif
}

// ---------------- fragment loaders (global) ----------------
// A fragment (16x32 f16, M x K): lane group g = lane>>4 holds contiguous
// K {kb+16g .. kb+16g+15} in halves 0..15 (same order as B).
__device__ __forceinline__ v16h ldA(const _Float16* __restrict__ base, int row,
                                    int stride, int kb, int ksel) {
  const _Float16* p = base + (size_t)row * stride + kb + ksel * 16;
  v8h lo = *(const v8h*)(p);
  v8h hi = *(const v8h*)(p + 8);
  v16h r;
#pragma unroll
  for (int i = 0; i < 8; ++i) { r[i] = lo[i]; r[i + 8] = hi[i]; }
  return r;
}

// B fragment (32x16 f16, K x N), weights stored (N,K) row-major: lane group g
// holds contiguous K {kb+16g .. kb+16g+15} for column n = lane&15.
__device__ __forceinline__ v16h ldB16(const _Float16* __restrict__ base, int nrow,
                                      int stride, int kb, int ksel) {
  const _Float16* p = base + (size_t)nrow * stride + kb + ksel * 16;
  v8h lo = *(const v8h*)(p);
  v8h hi = *(const v8h*)(p + 8);
  v16h r;
#pragma unroll
  for (int i = 0; i < 8; ++i) { r[i] = lo[i]; r[i + 8] = hi[i]; }
  return r;
}

// Same, but fp32 source weights -> convert in registers.
__device__ __forceinline__ v16h ldB32(const float* __restrict__ base, int nrow,
                                      int stride, int kb, int ksel) {
  const float* p = base + (size_t)nrow * stride + kb + ksel * 16;
  v4f f0 = *(const v4f*)(p);
  v4f f1 = *(const v4f*)(p + 4);
  v4f f2 = *(const v4f*)(p + 8);
  v4f f3 = *(const v4f*)(p + 12);
  v16h r;
#pragma unroll
  for (int i = 0; i < 4; ++i) {
    r[i]      = (_Float16)f0[i];
    r[4 + i]  = (_Float16)f1[i];
    r[8 + i]  = (_Float16)f2[i];
    r[12 + i] = (_Float16)f3[i];
  }
  return r;
}

// ---------------- fragment loaders (LDS, chunk = 128 rows x 32 halves) -------
__device__ __forceinline__ v16h ldA_lds(const _Float16* As, int m, int lane) {
  const _Float16* p = As + (m * 16 + (lane & 15)) * 32 + (lane >> 4) * 16;
  v8h lo = *(const v8h*)(p);
  v8h hi = *(const v8h*)(p + 8);
  v16h r;
#pragma unroll
  for (int i = 0; i < 8; ++i) { r[i] = lo[i]; r[i + 8] = hi[i]; }
  return r;
}

__device__ __forceinline__ v16h ldB_lds(const _Float16* Bs, int wv, int lane) {
  const _Float16* p = Bs + (wv * 16 + (lane & 15)) * 32 + (lane >> 4) * 16;
  v8h lo = *(const v8h*)(p);
  v8h hi = *(const v8h*)(p + 8);
  v16h r;
#pragma unroll
  for (int i = 0; i < 8; ++i) { r[i] = lo[i]; r[i + 8] = hi[i]; }
  return r;
}

// ---------------- staging kernels ----------------
// W_ih (8192 x 133 fp32) -> padded f16 (8192 x 160)
__global__ void k_convert_wih(const float* __restrict__ wih, _Float16* __restrict__ wih16) {
  int idx = blockIdx.x * 256 + threadIdx.x;            // 8192*160
  int n = idx / KX, k = idx - n * KX;
  wih16[idx] = (k < 133) ? (_Float16)wih[(size_t)n * 133 + k] : (_Float16)0.0f;
}

// W_hh (8192 x 2048 fp32) -> f16, 4 elements per thread
__global__ void k_convert_whh(const float* __restrict__ w, _Float16* __restrict__ w16) {
  size_t i = ((size_t)blockIdx.x * 256 + threadIdx.x) * 4;
  v4f f = *(const v4f*)(w + i);
  v4h h;
#pragma unroll
  for (int j = 0; j < 4; ++j) h[j] = (_Float16)f[j];
  *(v4h*)(w16 + i) = h;
}

// xbuf (128 x 160 f16): [ z(128) | stroke0(5) = (0,0,1,0,0) | zero pad ]
__global__ void k_init_x(const float* __restrict__ z, _Float16* __restrict__ xbuf) {
  int idx = blockIdx.x * 256 + threadIdx.x;            // 128*160
  int r = idx / KX, k = idx - r * KX;
  float v = 0.0f;
  if (k < LAT)            v = z[(size_t)r * LAT + k];
  else if (k == LAT + 2)  v = 1.0f;                    // pen-down one-hot
  xbuf[idx] = (_Float16)v;
}

// ---------------- fc_in: hc = tanh(z @ fc_in_w.T + b); h0 -> hbuf/f16, c0 -> cbuf
__global__ __launch_bounds__(256) void k_fcin(const _Float16* __restrict__ xbuf,
                                              const float* __restrict__ w,
                                              const float* __restrict__ bias,
                                              float* __restrict__ cbuf,
                                              _Float16* __restrict__ hbuf) {
  int wave = threadIdx.x >> 5, lane = threadIdx.x & 31;
  int mt = blockIdx.x >> 3;                            // 0..7
  int nbase = (blockIdx.x & 7) * 512 + wave * 64;      // N = 4096
  int arow = mt * 16 + (lane & 15);
  int ksel = lane >> 4;
  v8f acc[4] = {};
#pragma unroll
  for (int kb = 0; kb < LAT; kb += 32) {
    v16h a = ldA(xbuf, arow, KX, kb, ksel);
#pragma unroll
    for (int t = 0; t < 4; ++t) {
      v16h b = ldB32(w, nbase + t * 16 + (lane & 15), LAT, kb, ksel);
      acc[t] = WMMA_F16(a, b, acc[t]);
    }
  }
  int srow = mt * 16 + ksel * 8;                       // C layout: lane group -> M+8
#pragma unroll
  for (int t = 0; t < 4; ++t) {
    int n = nbase + t * 16 + (lane & 15);
#pragma unroll
    for (int r = 0; r < 8; ++r) {
      float v = tanhf(acc[t][r] + bias[n]);
      int m = srow + r;
      if (n < HID) {
        hbuf[(size_t)m * HID + n] = (_Float16)v;
      } else {
        cbuf[(size_t)m * HID + (n - HID)] = v;
      }
    }
  }
}

// ---------------- gates = [x|h] @ [W_ih|W_hh].T + biases (LDS-tiled) ---------
// One block owns a 128-wide N slab and all of M=128. Per 32-K step the block
// stages 8KB A + 8KB B into LDS (double-buffered). Staging uses the CDNA5
// async global->LDS engine when available (ASYNCcnt), overlapping the copy of
// chunk k+1 with the 8 WMMAs of chunk k. A and B are read from L2 exactly once
// per step.
template <bool WF16>
__global__ __launch_bounds__(256) void k_gates_t(
    const _Float16* __restrict__ xbuf, const _Float16* __restrict__ hbuf,
    const _Float16* __restrict__ wih16, const _Float16* __restrict__ whh16,
    const float* __restrict__ whh32, const float* __restrict__ b_ih,
    const float* __restrict__ b_hh, float* __restrict__ gates) {
  __shared__ _Float16 As[2][128 * 32];
  __shared__ _Float16 Bs[2][128 * 32];
  const int tid  = threadIdx.x;
  const int wave = tid >> 5, lane = tid & 31;
  const int nbase = blockIdx.x * 128;                  // 64 blocks cover N=8192
  const int ksel = lane >> 4;

  auto stage = [&](int buf, int kb) {
#pragma unroll
    for (int j = 0; j < 2; ++j) {
      int q = tid * 2 + j;                             // 0..511
      int row = q >> 2, seg = (q & 3) * 8;             // 8 halves per transfer
      cp16_g2l(hbuf + (size_t)row * HID + kb + seg, &As[buf][row * 32 + seg]);
      if (WF16) {
        cp16_g2l(whh16 + (size_t)(nbase + row) * HID + kb + seg,
                 &Bs[buf][row * 32 + seg]);
      } else {
        const float* p = whh32 + (size_t)(nbase + row) * HID + kb + seg;
        v4f f0 = *(const v4f*)(p);
        v4f f1 = *(const v4f*)(p + 4);
        v8h h;
#pragma unroll
        for (int i = 0; i < 4; ++i) { h[i] = (_Float16)f0[i]; h[4 + i] = (_Float16)f1[i]; }
        *(v8h*)(&Bs[buf][row * 32 + seg]) = h;
      }
    }
  };

  v8f acc[8] = {};
  stage(0, 0);
  int buf = 0;
  for (int kb = 0; kb < HID; kb += 32) {
    async_wait0();                                     // own async copies done
    __syncthreads();                                   // chunk visible block-wide
    if (kb + 32 < HID) stage(buf ^ 1, kb + 32);        // prefetch next chunk
    v16h bfrag = ldB_lds(Bs[buf], wave, lane);
#pragma unroll
    for (int m = 0; m < 8; ++m) {
      v16h afrag = ldA_lds(As[buf], m, lane);
      acc[m] = WMMA_F16(afrag, bfrag, acc[m]);
    }
    buf ^= 1;
  }

  // input part: K = 160 (z + stroke + pad), tiny -> direct loads
  const int nrow = nbase + wave * 16 + (lane & 15);
#pragma unroll
  for (int kb = 0; kb < KX; kb += 32) {
    v16h bfrag = ldB16(wih16, nrow, KX, kb, ksel);
#pragma unroll
    for (int m = 0; m < 8; ++m) {
      v16h afrag = ldA(xbuf, m * 16 + (lane & 15), KX, kb, ksel);
      acc[m] = WMMA_F16(afrag, bfrag, acc[m]);
    }
  }

  float bb = b_ih[nrow] + b_hh[nrow];
#pragma unroll
  for (int m = 0; m < 8; ++m) {
    int srow = m * 16 + ksel * 8;
#pragma unroll
    for (int r = 0; r < 8; ++r)
      gates[(size_t)(srow + r) * GATES + nrow] = acc[m][r] + bb;
  }
}

// ---------------- LSTM elementwise ----------------
__device__ __forceinline__ float sigm(float x) { return 1.0f / (1.0f + expf(-x)); }

__global__ void k_lstm(const float* __restrict__ gates, float* __restrict__ cbuf,
                       _Float16* __restrict__ hbuf) {
  int idx = blockIdx.x * 256 + threadIdx.x;            // 128*2048
  int row = idx >> 11, col = idx & (HID - 1);
  const float* g = gates + (size_t)row * GATES;
  float ig = g[col], fg = g[HID + col], gg = g[2 * HID + col], og = g[3 * HID + col];
  float cn = sigm(fg) * cbuf[idx] + sigm(ig) * tanhf(gg);
  float hn = sigm(og) * tanhf(cn);
  cbuf[idx] = cn;
  hbuf[idx] = (_Float16)hn;
}

// ---------------- proj: out = h @ proj_w.T + b (128 x 63, padded to 64) ------
__global__ __launch_bounds__(128) void k_proj(const _Float16* __restrict__ hbuf,
                                              const float* __restrict__ pw,
                                              const float* __restrict__ pb,
                                              float* __restrict__ obuf) {
  int wave = threadIdx.x >> 5, lane = threadIdx.x & 31;   // wave = N-tile 0..3
  int mt = blockIdx.x;                                    // 0..7
  int arow = mt * 16 + (lane & 15);
  int ksel = lane >> 4;
  int nrow = wave * 16 + (lane & 15);
  v8f acc = {};
  for (int kb = 0; kb < HID; kb += 32) {
    v16h a = ldA(hbuf, arow, HID, kb, ksel);
    v16h b;
    if (nrow < NOUT) b = ldB32(pw, nrow, HID, kb, ksel);
    else { v16h zz = {}; b = zz; }                        // padded row 63
    acc = WMMA_F16(a, b, acc);
  }
  int srow = mt * 16 + ksel * 8;
  if (nrow < NOUT) {
    float bb = pb[nrow];
#pragma unroll
    for (int r = 0; r < 8; ++r)
      obuf[(size_t)(srow + r) * NOUTP + nrow] = acc[r] + bb;
  }
}

// ---------------- threefry2x32 (JAX-compatible) ----------------
__device__ __forceinline__ unsigned rotl32(unsigned v, int s) {
  return (v << s) | (v >> (32 - s));
}

__device__ __forceinline__ void tf_block(unsigned k0, unsigned k1, unsigned c0,
                                         unsigned c1, unsigned& o0, unsigned& o1) {
  unsigned ks2 = k0 ^ k1 ^ 0x1BD11BDAu;
  unsigned x0 = c0 + k0, x1 = c1 + k1;
  const int rA[4] = {13, 15, 26, 6};
  const int rB[4] = {17, 29, 16, 24};
  auto grp = [&](const int* r, unsigned a0, unsigned a1, unsigned inc) {
#pragma unroll
    for (int i = 0; i < 4; ++i) { x0 += x1; x1 = rotl32(x1, r[i]); x1 ^= x0; }
    x0 += a0; x1 += a1 + inc;
  };
  grp(rA, k1, ks2, 1u);
  grp(rB, ks2, k0, 2u);
  grp(rA, k0, k1, 3u);
  grp(rB, k1, ks2, 4u);
  grp(rA, ks2, k0, 5u);
  o0 = x0; o1 = x1;
}

// random_bits(key, (128,)) element i: counters split as (0..63 | 64..127)
__device__ __forceinline__ unsigned draw_bits(unsigned k0, unsigned k1, int i) {
  unsigned o0, o1;
  if (i < 64) { tf_block(k0, k1, (unsigned)i, (unsigned)(i + 64), o0, o1); return o0; }
  tf_block(k0, k1, (unsigned)(i - 64), (unsigned)i, o0, o1);
  return o1;
}

__device__ __forceinline__ float u01(unsigned bits) {
  return __uint_as_float((bits >> 9) | 0x3f800000u) - 1.0f;
}

__device__ __forceinline__ float jax_normal(unsigned bits) {
  const float lo = -0.99999994f;                        // nextafter(-1, 0)
  float u = u01(bits) * (1.0f - lo) + lo;
  u = fmaxf(lo, u);
  return 1.41421356237309515f * erfinvf(u);
}

// ---------------- GMM sampling, one thread per batch row ----------------
__global__ void k_sample(const float* __restrict__ obuf, const int* __restrict__ Ns,
                         int step, float* __restrict__ strokes,
                         _Float16* __restrict__ xbuf) {
  int i = threadIdx.x;
  if (i >= B_SZ) return;
  const float* o = obuf + (size_t)i * NOUTP;

  // per-step key: split(key(42), 10) -> block((0,42), (step, step+10))
  unsigned kj0, kj1; tf_block(0u, 42u, (unsigned)step, (unsigned)(step + 10), kj0, kj1);
  unsigned k10, k11; tf_block(kj0, kj1, 0u, 4u, k10, k11);
  unsigned k20, k21; tf_block(kj0, kj1, 1u, 5u, k20, k21);
  unsigned k30, k31; tf_block(kj0, kj1, 2u, 6u, k30, k31);
  unsigned k40, k41; tf_block(kj0, kj1, 3u, 7u, k40, k41);

  float u  = fmaxf(0.0f, u01(draw_bits(k10, k11, i)));
  float rx = jax_normal(draw_bits(k20, k21, i));
  float ry = jax_normal(draw_bits(k30, k31, i));
  float u2 = fmaxf(0.0f, u01(draw_bits(k40, k41, i)));

  // mixture softmax over pi logits (T = 0.5 -> *2)
  float lg[10], mxl = -1e30f;
#pragma unroll
  for (int m = 0; m < 10; ++m) { lg[m] = o[6 * m] * 2.0f; mxl = fmaxf(mxl, lg[m]); }
  float s = 0.0f;
#pragma unroll
  for (int m = 0; m < 10; ++m) { lg[m] = expf(lg[m] - mxl); s += lg[m]; }
  float inv = 1.0f / s, cs = 0.0f;
  int idx = 0;
#pragma unroll
  for (int m = 0; m < 10; ++m) { cs += lg[m] * inv; idx += (cs < u) ? 1 : 0; }
  if (idx > 9) idx = 9;

  float mx  = o[6 * idx + 1];
  float my  = o[6 * idx + 2];
  float sx  = expf(o[6 * idx + 3]) * SQRT_T;
  float sy  = expf(o[6 * idx + 4]) * SQRT_T;
  float rho = tanhf(o[6 * idx + 5]);
  float ox = mx + sx * rx;
  float oy = my + sy * (rho * ox + sqrtf(fmaxf(0.0f, 1.0f - rho * rho)) * ry);

  // pen softmax (3-way)
  float pl[3], mp = -1e30f;
#pragma unroll
  for (int m = 0; m < 3; ++m) { pl[m] = o[60 + m] * 2.0f; mp = fmaxf(mp, pl[m]); }
  float ps = 0.0f;
#pragma unroll
  for (int m = 0; m < 3; ++m) { pl[m] = expf(pl[m] - mp); ps += pl[m]; }
  float pinv = 1.0f / ps, pcs = 0.0f;
  int pidx = 0;
#pragma unroll
  for (int m = 0; m < 3; ++m) { pcs += pl[m] * pinv; pidx += (pcs < u2) ? 1 : 0; }
  if (pidx > 2) pidx = 2;

  float v0 = ox, v1 = oy;
  float v2 = (pidx == 0) ? 1.0f : 0.0f;
  float v3 = (pidx == 1) ? 1.0f : 0.0f;
  float v4 = (pidx == 2) ? 1.0f : 0.0f;
  if (step > Ns[i]) { v0 = v1 = v2 = v3 = 0.0f; v4 = 1.0f; }  // "empty" stroke

  float* d = strokes + (size_t)step * B_SZ * 5 + (size_t)i * 5;
  d[0] = v0; d[1] = v1; d[2] = v2; d[3] = v3; d[4] = v4;

  _Float16* xr = xbuf + (size_t)i * KX + LAT;           // next-step input cols 128..132
  xr[0] = (_Float16)v0; xr[1] = (_Float16)v1; xr[2] = (_Float16)v2;
  xr[3] = (_Float16)v3; xr[4] = (_Float16)v4;
}

// ---------------- launch ----------------
extern "C" void kernel_launch(void* const* d_in, const int* in_sizes, int n_in,
                              void* d_out, int out_size, void* d_ws, size_t ws_size,
                              hipStream_t stream) {
  const float* z       = (const float*)d_in[0];
  const int*   N_s     = (const int*)d_in[1];
  const float* fc_in_w = (const float*)d_in[2];
  const float* fc_in_b = (const float*)d_in[3];
  const float* W_ih    = (const float*)d_in[4];
  const float* W_hh    = (const float*)d_in[5];
  const float* b_ih    = (const float*)d_in[6];
  const float* b_hh    = (const float*)d_in[7];
  const float* proj_w  = (const float*)d_in[8];
  const float* proj_b  = (const float*)d_in[9];
  float* strokes = (float*)d_out;

  char* ws = (char*)d_ws;
  _Float16* xbuf  = (_Float16*)(ws);                       //   40,960 B
  _Float16* hbuf  = (_Float16*)(ws + 40960);               //  524,288 B
  _Float16* wih16 = (_Float16*)(ws + 565248);              // 2,621,440 B
  float*    gates = (float*)   (ws + 3186688);             // 4,194,304 B
  float*    cbuf  = (float*)   (ws + 7380992);             // 1,048,576 B
  float*    obuf  = (float*)   (ws + 8429568);             //    32,768 B
  const size_t base_need = 8462336;                        // ~8.5 MB
  const size_t whh16_bytes = (size_t)GATES * HID * sizeof(_Float16);  // 32 MB
  const bool staged = ws_size >= base_need + whh16_bytes;  // fixed per environment
  _Float16* whh16 = (_Float16*)(ws + base_need);

  k_convert_wih<<<(GATES * KX) / 256, 256, 0, stream>>>(W_ih, wih16);
  if (staged)
    k_convert_whh<<<(GATES * HID / 4) / 256, 256, 0, stream>>>(W_hh, whh16);
  k_init_x<<<(B_SZ * KX) / 256, 256, 0, stream>>>(z, xbuf);
  k_fcin<<<64, 256, 0, stream>>>(xbuf, fc_in_w, fc_in_b, cbuf, hbuf);

  for (int step = 0; step < NMAX; ++step) {
    if (staged)
      k_gates_t<true><<<64, 256, 0, stream>>>(xbuf, hbuf, wih16, whh16, W_hh,
                                              b_ih, b_hh, gates);
    else
      k_gates_t<false><<<64, 256, 0, stream>>>(xbuf, hbuf, wih16, nullptr, W_hh,
                                               b_ih, b_hh, gates);
    k_lstm<<<(B_SZ * HID) / 256, 256, 0, stream>>>(gates, cbuf, hbuf);
    k_proj<<<8, 128, 0, stream>>>(hbuf, proj_w, proj_b, obuf);
    k_sample<<<1, 128, 0, stream>>>(obuf, N_s, step, strokes, xbuf);
  }
}
